// AttentionSelf_20100446945357
// MI455X (gfx1250) — matrix operand order repaired and checked by reading, not compiled, not run on hardware
//
#include <hip/hip_runtime.h>
#include <hip/hip_bf16.h>
#include <math.h>

typedef __bf16 bf16_t;
typedef __attribute__((ext_vector_type(16))) __bf16 v16bf;
typedef __attribute__((ext_vector_type(8)))  float  v8f;
typedef __attribute__((ext_vector_type(4)))  unsigned int u32x4;
typedef __attribute__((ext_vector_type(8)))  int i32x8;
typedef __attribute__((ext_vector_type(4)))  int i32x4;

__device__ __forceinline__ v8f wmma_bf16(v16bf a, v16bf b, v8f c) {
  // D = A(16x32 bf16) x B(32x16 bf16) + C(16x16 f32)
  return __builtin_amdgcn_wmma_f32_16x16x32_bf16(false, a, false, b, (short)0, c,
                                                 false, false);
}

__device__ __forceinline__ int clampi(int v, int lo, int hi) {
  return v < lo ? lo : (v > hi ? hi : v);
}

// ---------------- WMMA fragment loaders (contiguous k = 16*half + j order) ----------------

// A fragment: bf16 row-major [M x K], tile (m0,k0).
__device__ __forceinline__ v16bf load_a_bf16(const bf16_t* A, int lda, int m0,
                                             int k0, int lane) {
  int r = lane & 15, hf = lane >> 4;
  const bf16_t* p0 = A + (size_t)(m0 + r) * lda + k0 + hf * 16;
  v16bf a;
#pragma unroll
  for (int j = 0; j < 16; ++j) a[j] = p0[j];
  return a;
}

// B fragment from f32 row-major [K x N], tile (k0,n0): lane holds column n0+r,
// slot j -> K = k0 + half*16 + j (converted to bf16 on the fly).
__device__ __forceinline__ v16bf load_b_f32(const float* B, int ldb, int k0,
                                            int n0, int lane) {
  int r = lane & 15, hf = lane >> 4;
  const float* p = B + (size_t)(k0 + hf * 16) * ldb + n0 + r;
  v16bf b;
#pragma unroll
  for (int j = 0; j < 16; ++j) b[j] = (bf16_t)p[(size_t)j * ldb];
  return b;
}

// B fragment from bf16 stored K-contiguous per column: Bt[n * ldt + k].
__device__ __forceinline__ v16bf load_b_tr(const bf16_t* Bt, int ldt, int k0,
                                           int n0, int lane) {
  int r = lane & 15, hf = lane >> 4;
  const bf16_t* p = Bt + (size_t)(n0 + r) * ldt + k0 + hf * 16;
  v16bf b;
#pragma unroll
  for (int j = 0; j < 16; ++j) b[j] = p[j];
  return b;
}

// ---------------- TDM: DMA a contiguous 32KB region into LDS ----------------
// D# per ISA 8.3/8.4: 2D tensor, data_size=2B, 8192x2 elements, type=2.
// This toolchain exposes the 6-arg builtin:
// (u32x4 g0, i32x8 g1, i32x4 g2, i32x4 g3, i32x8 extra, i32 cpol)
__device__ __forceinline__ void tdm_load_32kb(unsigned lds_addr,
                                              const void* gptr) {
#if __has_builtin(__builtin_amdgcn_tensor_load_to_lds)
  unsigned long long ga = (unsigned long long)gptr;
  u32x4 g0;
  g0[0] = 1u;                                    // count=1 (valid user D#)
  g0[1] = lds_addr;                              // LDS byte address
  g0[2] = (unsigned)(ga & 0xffffffffu);          // global_addr[31:0]
  g0[3] = (unsigned)((ga >> 32) & 0x01ffffffu)   // global_addr[56:32]
          | 0x80000000u;                         // type=2 ("image")
  i32x8 g1;
  g1[0] = (int)(1u << 16);                       // wg_mask=0, data_size=1 (2B)
  g1[1] = (int)((8192u & 0xffffu) << 16);        // tensor_dim0 lo -> [63:48]
  g1[2] = (int)((8192u >> 16) | (2u << 16));     // tensor_dim0 hi, tensor_dim1 lo
  g1[3] = (int)((8192u & 0xffffu) << 16);        // tensor_dim1 hi=0, tile_dim0
  g1[4] = (int)2u;                               // tile_dim1=2, tile_dim2=0
  g1[5] = (int)8192u;                            // tensor_dim0_stride lo
  g1[6] = 0;
  g1[7] = 0;
  i32x4 z4 = {0, 0, 0, 0};
  i32x8 z8 = {0, 0, 0, 0, 0, 0, 0, 0};
  __builtin_amdgcn_tensor_load_to_lds(g0, g1, z4, z4, z8, 0);
#else
  (void)lds_addr; (void)gptr;
#endif
}

// ---------------- weight f32 -> bf16 ----------------
__global__ __launch_bounds__(256) void cvt_bf16_kernel(const float* __restrict__ s,
                                                       bf16_t* __restrict__ d,
                                                       int n) {
  int i = blockIdx.x * 256 + threadIdx.x;
  if (i < n) d[i] = (bf16_t)s[i];
}

// ---------------- generic 256xK channel-projection GEMM ----------------
// out[o,n] = sum_c W[o,c] * X[c,n] + bias[o], per batch b (blockIdx.y).
// MODE 0: q-proj   -> outf = q f32 [b,256,4096], outb = qT bf16 [bh,4096,64]
// MODE 1: k-proj   -> outb = kT bf16 [bh,256n,64c]
// MODE 2: v-proj   -> outb = v  bf16 [bh,64c,256n]
// MODE 3: out-proj -> outf = y f32 [b,256,4096]
template <int MODE>
__global__ __launch_bounds__(256) void proj_kernel(
    const bf16_t* __restrict__ W, const float* __restrict__ bias,
    const float* __restrict__ X_all, int ldx, size_t xstride,
    float* __restrict__ outf, bf16_t* __restrict__ outb) {
  int b = blockIdx.y;
  int n_base = blockIdx.x * 64;
  int lane = threadIdx.x & 31, w = threadIdx.x >> 5;
  int r = lane & 15, hf = lane >> 4;
  const float* X = X_all + (size_t)b * xstride;

  v8f acc[2][4] = {};
#pragma unroll 1
  for (int k0 = 0; k0 < 256; k0 += 32) {
    if (k0 < 224) {  // prefetch next K-step of the B operand (global_prefetch_b8)
      __builtin_prefetch(X + (size_t)(k0 + 32 + hf * 16) * ldx + n_base + r, 0, 1);
      __builtin_prefetch(X + (size_t)(k0 + 40 + hf * 16) * ldx + n_base + r, 0, 1);
    }
    v16bf a0 = load_a_bf16(W, 256, (w * 2 + 0) * 16, k0, lane);
    v16bf a1 = load_a_bf16(W, 256, (w * 2 + 1) * 16, k0, lane);
#pragma unroll
    for (int t = 0; t < 4; ++t) {
      v16bf bb = load_b_f32(X, ldx, k0, n_base + t * 16, lane);
      acc[0][t] = wmma_bf16(a0, bb, acc[0][t]);
      acc[1][t] = wmma_bf16(a1, bb, acc[1][t]);
    }
  }

#pragma unroll
  for (int mi = 0; mi < 2; ++mi) {
    int m0 = (w * 2 + mi) * 16;
#pragma unroll
    for (int t = 0; t < 4; ++t) {
      int n = n_base + t * 16 + r;
#pragma unroll
      for (int i = 0; i < 8; ++i) {
        int o = m0 + hf * 8 + i;
        float v = acc[mi][t][i] + bias[o];
        if (MODE == 0) {
          outf[((size_t)b * 256 + o) * 4096 + n] = v;
          outb[(((size_t)b * 4 + (o >> 6)) * 4096 + n) * 64 + (o & 63)] =
              (bf16_t)v;
        } else if (MODE == 1) {
          outb[(((size_t)b * 4 + (o >> 6)) * 256 + n) * 64 + (o & 63)] =
              (bf16_t)v;
        } else if (MODE == 2) {
          outb[(((size_t)b * 4 + (o >> 6)) * 64 + (o & 63)) * 256 + n] =
              (bf16_t)v;
        } else {
          outf[((size_t)b * 256 + o) * 4096 + n] = v;
        }
      }
    }
  }
}

// ---------------- offset branch: dwconv4x4/s4 + LN + GELU + pointwise ------
__global__ __launch_bounds__(256) void offset_kernel(
    const float* __restrict__ qf, const float* __restrict__ dw_w,
    const float* __restrict__ dw_b, const float* __restrict__ ln_g,
    const float* __restrict__ ln_b, const float* __restrict__ pw_w,
    float* __restrict__ pos) {
  __shared__ float red[256];
  int cell = blockIdx.x;            // b*256 + (ky*16 + kx)
  int b = cell >> 8, cidx = cell & 255;
  int ky = cidx >> 4, kx = cidx & 15;
  int c = threadIdx.x;

  const float* qb = qf + ((size_t)b * 256 + c) * 4096;
  float s = 0.f;
#pragma unroll
  for (int p = 0; p < 4; ++p)
#pragma unroll
    for (int q = 0; q < 4; ++q)
      s += qb[(ky * 4 + p) * 64 + kx * 4 + q] * dw_w[c * 16 + p * 4 + q];
  s += dw_b[c];

  auto block_sum = [&](float v) -> float {
    red[threadIdx.x] = v;
    __syncthreads();
    for (int st = 128; st > 0; st >>= 1) {
      if (threadIdx.x < st) red[threadIdx.x] += red[threadIdx.x + st];
      __syncthreads();
    }
    float rr = red[0];
    __syncthreads();
    return rr;
  };

  float mu = block_sum(s) * (1.f / 256.f);
  float var = block_sum((s - mu) * (s - mu)) * (1.f / 256.f);
  float t = (s - mu) * rsqrtf(var + 1e-5f) * ln_g[c] + ln_b[c];
  float g = 0.5f * t * (1.f + erff(t * 0.70710678118654752f));  // exact GELU

  float offy = block_sum(pw_w[c] * g);
  float offx = block_sum(pw_w[256 + c] * g);

  if (threadIdx.x == 0) {
    float refy = (0.5f + (float)ky) * (2.f / 15.f) - 1.f;
    float refx = (0.5f + (float)kx) * (2.f / 15.f) - 1.f;
    float py = fminf(fmaxf(offy + refy, -1.f), 1.f);
    float px = fminf(fmaxf(offx + refx, -1.f), 1.f);
    pos[(size_t)cell * 2 + 0] = py;
    pos[(size_t)cell * 2 + 1] = px;
  }
}

// ---------------- deformable bilinear gather: xs[b,c,n] ----------------
__global__ __launch_bounds__(256) void gather_kernel(const float* __restrict__ x0,
                                                     const float* __restrict__ pos,
                                                     float* __restrict__ xs) {
  int bn = blockIdx.x;  // b*256 + n
  int b = bn >> 8, n = bn & 255;
  float py = pos[(size_t)bn * 2 + 0], px = pos[(size_t)bn * 2 + 1];
  float fpx = (px + 1.f) * 0.5f * 63.f;
  float fpy = (py + 1.f) * 0.5f * 63.f;
  float fx = floorf(fpx), fy = floorf(fpy);
  float wx = fpx - fx, wy = fpy - fy;
  int x0i = clampi((int)fx, 0, 63), x1i = clampi((int)fx + 1, 0, 63);
  int y0i = clampi((int)fy, 0, 63), y1i = clampi((int)fy + 1, 0, 63);
  int c = threadIdx.x;
  const float* base = x0 + ((size_t)b * 512 + c) * 4096;  // q = x0[:, :256]
  float v = base[y0i * 64 + x0i] * (1.f - wx) * (1.f - wy) +
            base[y0i * 64 + x1i] * wx * (1.f - wy) +
            base[y1i * 64 + x0i] * (1.f - wx) * wy +
            base[y1i * 64 + x1i] * wx * wy;
  xs[((size_t)b * 256 + c) * 256 + n] = v;
}

// ---------------- fused attention: S=QtK*scale+RPE, softmax, O=PVt --------
#define RPE_N 16129  // 127*127
__global__ __launch_bounds__(256) void attn_kernel(
    const bf16_t* __restrict__ qt, const bf16_t* __restrict__ kt,
    const bf16_t* __restrict__ vt, const float* __restrict__ pos,
    const float* __restrict__ rpe, float* __restrict__ out) {
  __shared__ float  s_rpe[RPE_N];
  __shared__ float  s_posx[256];
  __shared__ float  s_posy[256];
  __shared__ bf16_t s_p[8 * 16 * 256];  // per-wave 16x256 P tile, bf16
  __shared__ bf16_t s_k[256 * 64];      // K tile  [n][c] (32 KB, TDM-loaded)
  __shared__ bf16_t s_v[64 * 256];      // V tile  [c][n] (32 KB, TDM-loaded)

  int tid = threadIdx.x;
  int bh = blockIdx.y;  // b*4 + h
  int b = bh >> 2, h = bh & 3;
  int m_base = blockIdx.x * 128;
  int lane = tid & 31, w = tid >> 5;
  int r = lane & 15, hf = lane >> 4;
  int m0 = m_base + w * 16;

  const bf16_t* qtb = qt + (size_t)bh * 4096 * 64;  // [m][c]
  const bf16_t* ktb = kt + (size_t)bh * 256 * 64;   // [n][c]
  const bf16_t* vtb = vt + (size_t)bh * 64 * 256;   // [c][n]

#if __has_builtin(__builtin_amdgcn_tensor_load_to_lds)
  if (w == 0) {  // one wave drives the Tensor Data Mover for K and V tiles
    tdm_load_32kb((unsigned)(size_t)&s_k[0], ktb);
    tdm_load_32kb((unsigned)(size_t)&s_v[0], vtb);
    __builtin_amdgcn_s_wait_tensorcnt(0);
  }
#else
  for (int i = tid; i < 256 * 64; i += 256) { s_k[i] = ktb[i]; s_v[i] = vtb[i]; }
#endif
  for (int i = tid; i < RPE_N; i += 256) s_rpe[i] = rpe[(size_t)h * RPE_N + i];
  s_posy[tid] = pos[(size_t)(b * 256 + tid) * 2 + 0];
  s_posx[tid] = pos[(size_t)(b * 256 + tid) * 2 + 1];
  __syncthreads();

  // ---- S = Q^T K : M=16(rows m), N=256(n), K=64(c); B operand from LDS ----
  v16bf a0 = load_a_bf16(qtb, 64, m0, 0, lane);
  v16bf a1 = load_a_bf16(qtb, 64, m0, 32, lane);
  v8f acc[16] = {};
#pragma unroll
  for (int t = 0; t < 16; ++t) {
    v16bf b0 = load_b_tr(s_k, 64, 0, t * 16, lane);
    v16bf b1 = load_b_tr(s_k, 64, 32, t * 16, lane);
    acc[t] = wmma_bf16(a0, b0, acc[t]);
    acc[t] = wmma_bf16(a1, b1, acc[t]);
  }

  // ---- RPE bias + softmax (rows spread over 16 lanes of one half) ----
  bf16_t* sp = s_p + (size_t)w * (16 * 256);
  const float scale = 0.125f;  // hc^-0.5
#pragma unroll
  for (int i = 0; i < 8; ++i) {
    int m = m0 + hf * 8 + i;
    float qy = (float)(m >> 6) * (2.f / 63.f) - 1.f;
    float qx = (float)(m & 63) * (2.f / 63.f) - 1.f;
    float mx = -3.4e38f;
#pragma unroll
    for (int t = 0; t < 16; ++t) {
      int n = t * 16 + r;
      float gx = (qx - s_posx[n]) * 0.5f;
      float gy = (qy - s_posy[n]) * 0.5f;
      float fpx = (gx + 1.f) * 63.f;  // (g+1)*0.5*(127-1)
      float fpy = (gy + 1.f) * 63.f;
      float fx = floorf(fpx), fy = floorf(fpy);
      float wx = fpx - fx, wy = fpy - fy;
      int x0i = clampi((int)fx, 0, 126), x1i = clampi((int)fx + 1, 0, 126);
      int y0i = clampi((int)fy, 0, 126), y1i = clampi((int)fy + 1, 0, 126);
      float bias = s_rpe[y0i * 127 + x0i] * (1.f - wx) * (1.f - wy) +
                   s_rpe[y0i * 127 + x1i] * wx * (1.f - wy) +
                   s_rpe[y1i * 127 + x0i] * (1.f - wx) * wy +
                   s_rpe[y1i * 127 + x1i] * wx * wy;
      float sv = acc[t][i] * scale + bias;
      acc[t][i] = sv;
      mx = fmaxf(mx, sv);
    }
#pragma unroll
    for (int st = 1; st < 16; st <<= 1) mx = fmaxf(mx, __shfl_xor(mx, st, 32));
    float sum = 0.f;
#pragma unroll
    for (int t = 0; t < 16; ++t) {
      float p = __expf(acc[t][i] - mx);
      acc[t][i] = p;
      sum += p;
    }
#pragma unroll
    for (int st = 1; st < 16; st <<= 1) sum += __shfl_xor(sum, st, 32);
    float inv = 1.f / sum;
#pragma unroll
    for (int t = 0; t < 16; ++t)
      sp[(hf * 8 + i) * 256 + t * 16 + r] = (bf16_t)(acc[t][i] * inv);
  }
  __syncthreads();

  // ---- O = P V^T : M=16(m), N=64(c), K=256(n); both operands from LDS ----
  v8f oacc[4] = {};
#pragma unroll 1
  for (int k0 = 0; k0 < 256; k0 += 32) {
    v16bf a = load_a_bf16(sp, 256, 0, k0, lane);
#pragma unroll
    for (int t = 0; t < 4; ++t) {
      v16bf bb = load_b_tr(s_v, 256, k0, t * 16, lane);
      oacc[t] = wmma_bf16(a, bb, oacc[t]);
    }
  }
#pragma unroll
  for (int t = 0; t < 4; ++t) {
#pragma unroll
    for (int i = 0; i < 8; ++i) {
      int c = t * 16 + r;
      int m = m0 + hf * 8 + i;
      out[((size_t)b * 256 + h * 64 + c) * 4096 + m] = oacc[t][i];
    }
  }
}

// ---------------- epilogue: res = x0*exp(s)+t, jac = sum(s) per batch -----
__global__ void zero_jac_kernel(float* jac) {
  if (threadIdx.x < 8) jac[threadIdx.x] = 0.f;
}

__global__ __launch_bounds__(256) void combine_kernel(
    const float* __restrict__ x0, const float* __restrict__ sbuf,
    const float* __restrict__ tbuf, float* __restrict__ res,
    float* __restrict__ jac) {
  __shared__ float red[256];
  size_t idx = (size_t)blockIdx.x * 256 + threadIdx.x;
  int b = (int)(idx >> 21);              // 512*4096 = 2^21 per batch
  int rem = (int)(idx & ((1u << 21) - 1));
  int cfull = rem >> 12;
  int m = rem & 4095;
  float xv = x0[idx];
  float sval = 0.f, outv;
  if (cfull < 256) {
    outv = xv;  // zero scale / zero translate channels
  } else {
    size_t si = ((size_t)b * 256 + (cfull - 256)) * 4096 + m;
    sval = sbuf[si];
    outv = xv * expf(sval) + tbuf[si];
  }
  res[idx] = outv;

  red[threadIdx.x] = sval;
  __syncthreads();
  for (int st = 128; st > 0; st >>= 1) {
    if (threadIdx.x < st) red[threadIdx.x] += red[threadIdx.x + st];
    __syncthreads();
  }
  if (threadIdx.x == 0) atomicAdd(&jac[b], red[0]);
}

// ---------------- host launcher ----------------
extern "C" void kernel_launch(void* const* d_in, const int* in_sizes, int n_in,
                              void* d_out, int out_size, void* d_ws,
                              size_t ws_size, hipStream_t stream) {
  const float* x0 = (const float*)d_in[0];
  char* ws = (char*)d_ws;
  size_t off = 0;
  auto alloc = [&](size_t bytes) -> char* {
    char* p = ws + off;
    off = (off + bytes + 255) & ~(size_t)255;
    return p;
  };
  float*  qf     = (float*)alloc((size_t)8 * 256 * 4096 * 4);   // 32 MB
  bf16_t* qt     = (bf16_t*)alloc((size_t)8 * 256 * 4096 * 2);  // 16 MB
  float*  pos    = (float*)alloc((size_t)8 * 256 * 2 * 4);
  float*  xs     = (float*)alloc((size_t)8 * 256 * 256 * 4);    // 2 MB
  bf16_t* kt     = (bf16_t*)alloc((size_t)8 * 256 * 256 * 2);   // 1 MB
  bf16_t* vt     = (bf16_t*)alloc((size_t)8 * 256 * 256 * 2);   // 1 MB
  float*  attn_o = (float*)alloc((size_t)8 * 256 * 4096 * 4);   // 32 MB
  bf16_t* wqb    = (bf16_t*)alloc(65536 * 2);
  bf16_t* wkb    = (bf16_t*)alloc(65536 * 2);
  bf16_t* wvb    = (bf16_t*)alloc(65536 * 2);
  bf16_t* wob    = (bf16_t*)alloc(65536 * 2);
  float*  sbuf   = (float*)alloc((size_t)8 * 256 * 4096 * 4);   // 32 MB
  float*  tbuf   = (float*)alloc((size_t)8 * 256 * 4096 * 4);   // 32 MB

  float* res = (float*)d_out;
  float* jac = res + (size_t)8 * 512 * 4096;

  for (int blk = 0; blk < 2; ++blk) {
    int base = 1 + blk * 14;
    const float* dw_w = (const float*)d_in[base + 0];
    const float* dw_b = (const float*)d_in[base + 1];
    const float* ln_g = (const float*)d_in[base + 2];
    const float* ln_b = (const float*)d_in[base + 3];
    const float* pw_w = (const float*)d_in[base + 4];
    const float* wq   = (const float*)d_in[base + 5];
    const float* bq   = (const float*)d_in[base + 6];
    const float* wk   = (const float*)d_in[base + 7];
    const float* bk   = (const float*)d_in[base + 8];
    const float* wv   = (const float*)d_in[base + 9];
    const float* bv   = (const float*)d_in[base + 10];
    const float* wo   = (const float*)d_in[base + 11];
    const float* bo   = (const float*)d_in[base + 12];
    const float* rpe  = (const float*)d_in[base + 13];
    float* ybuf = blk ? tbuf : sbuf;

    cvt_bf16_kernel<<<256, 256, 0, stream>>>(wq, wqb, 65536);
    cvt_bf16_kernel<<<256, 256, 0, stream>>>(wk, wkb, 65536);
    cvt_bf16_kernel<<<256, 256, 0, stream>>>(wv, wvb, 65536);
    cvt_bf16_kernel<<<256, 256, 0, stream>>>(wo, wob, 65536);

    // q = wq x + bq (writes f32 q and bf16 qT)
    proj_kernel<0><<<dim3(64, 8), 256, 0, stream>>>(
        wqb, bq, x0, 4096, (size_t)512 * 4096, qf, qt);
    // offsets -> sampling positions
    offset_kernel<<<2048, 256, 0, stream>>>(qf, dw_w, dw_b, ln_g, ln_b, pw_w,
                                            pos);
    // deformable sampling
    gather_kernel<<<2048, 256, 0, stream>>>(x0, pos, xs);
    // k = wk xs + bk ; v = wv xs + bv
    proj_kernel<1><<<dim3(4, 8), 256, 0, stream>>>(
        wkb, bk, xs, 256, (size_t)256 * 256, nullptr, kt);
    proj_kernel<2><<<dim3(4, 8), 256, 0, stream>>>(
        wvb, bv, xs, 256, (size_t)256 * 256, nullptr, vt);
    // fused attention (TDM-staged K/V, LDS-resident RPE)
    attn_kernel<<<dim3(32, 32), 256, 0, stream>>>(qt, kt, vt, pos, rpe, attn_o);
    // final projection
    proj_kernel<3><<<dim3(64, 8), 256, 0, stream>>>(
        wob, bo, attn_o, 4096, (size_t)256 * 4096, ybuf, nullptr);
  }

  zero_jac_kernel<<<1, 32, 0, stream>>>(jac);
  combine_kernel<<<65536, 256, 0, stream>>>(x0, sbuf, tbuf, res, jac);
}
